// SuperGlueAttentionalGNN_81544249082309
// MI455X (gfx1250) — matrix operand order repaired and checked by reading, not compiled, not run on hardware
//
#include <hip/hip_runtime.h>
#include <hip/hip_bf16.h>

// ---------------------------------------------------------------------------
// SuperGlue attentional GNN forward, CDNA5 (gfx1250) WMMA implementation.
// L=18, B=2, N=1024, D=256, H=4, DH=64.  Rows M = B*N = 2048.
// All matmuls run on v_wmma_f32_16x16x32_f16 (f16 in, f32 accumulate).
// Weights are pre-converted once to f16, transposed (Wt[n][k]), so GEMM B
// tiles are staged with GLOBAL_LOAD_ASYNC_TO_LDS_B128 (ASYNCcnt) -- no cvt,
// no VGPR round trip.  Activations (A side) stay f32 and are staged via
// float4 loads + v_cvt_pk_f16_f32 + ds_store_b64, software-pipelined through
// double-buffered LDS.
// ---------------------------------------------------------------------------

typedef _Float16 v16h __attribute__((ext_vector_type(16)));
typedef float    v8f  __attribute__((ext_vector_type(8)));
typedef _Float16 h4   __attribute__((ext_vector_type(4)));
typedef _Float16 h2   __attribute__((ext_vector_type(2)));

#define NROWS   2048        // B*N
#define DMODEL  256
#define DHID    512
#define NSEQ    1024
#define NHEAD   4
#define DHEAD   64

#define APAD    40          // LDS row stride (halves) for A: 128x32 tile
#define BPAD    40          // LDS row stride (halves) for Bt: 64x32 tile (transposed)

// ---------------- async copy (global f16 -> LDS), ISA 15.18.3 ---------------

__device__ __forceinline__ void async_copy_b128(void* lds_dst, const void* gsrc) {
  uint32_t lds = (uint32_t)(uintptr_t)lds_dst;   // low 32 bits = LDS offset
  asm volatile("global_load_async_to_lds_b128 %0, %1, off"
               :: "v"(lds), "v"(gsrc)
               : "memory");
}
__device__ __forceinline__ void wait_async() {
  asm volatile("s_wait_asynccnt 0x0" ::: "memory");
}

// ---------------- WMMA fragment loaders (ISA 7.12.2 layouts) ----------------

// A matrix 16x32 f16: lane L(0..15) holds row M=L, K = {0..7,16..23};
// lane L+16 holds row M=L, K = {8..15,24..31}.  Two 16B contiguous chunks.
__device__ __forceinline__ v16h load_a_frag(const _Float16* p) {
  const int lane = threadIdx.x & 31;
  const _Float16* r = p + (lane & 15) * APAD + (lane >> 4) * 8;
  v16h a;
#pragma unroll
  for (int i = 0; i < 8; ++i) a[i] = r[i];
#pragma unroll
  for (int i = 0; i < 8; ++i) a[8 + i] = r[16 + i];
  return a;
}

// B matrix 32x16 f16 stored transposed in LDS as Bt[n][k]:
// lane n(0..15) holds col N=n, K = {0..7,16..23}; lane n+16 holds col N=n, K = {8..15,24..31}.
__device__ __forceinline__ v16h load_bt_frag(const _Float16* p) {
  const int lane = threadIdx.x & 31;
  const _Float16* r = p + (lane & 15) * BPAD + (lane >> 4) * 8;
  v16h b;
#pragma unroll
  for (int i = 0; i < 8; ++i) b[i] = r[i];
#pragma unroll
  for (int i = 0; i < 8; ++i) b[8 + i] = r[16 + i];
  return b;
}

// C/D 16x16 f32: lane holds N = lane&15; VGPR r holds M = r + 8*(lane>=16).

__device__ __forceinline__ h4 cvt4(float4 f) {
  h4 h = {(_Float16)f.x, (_Float16)f.y, (_Float16)f.z, (_Float16)f.w};
  return h;
}

// Store a 2(k) x 4(n) sub-block into transposed Bt: 4 packed h2 (ds_store_b32).
__device__ __forceinline__ void store_bpair(_Float16* p /*&Bt[c][r]*/, float4 f0, float4 f1) {
  h2 t0 = {(_Float16)f0.x, (_Float16)f1.x};
  h2 t1 = {(_Float16)f0.y, (_Float16)f1.y};
  h2 t2 = {(_Float16)f0.z, (_Float16)f1.z};
  h2 t3 = {(_Float16)f0.w, (_Float16)f1.w};
  *(h2*)(p)            = t0;
  *(h2*)(p + BPAD)     = t1;
  *(h2*)(p + 2 * BPAD) = t2;
  *(h2*)(p + 3 * BPAD) = t3;
}

// ---------------- weight convert+transpose: W[K][N] f32 -> Wt[N][K] f16 -----

__global__ void convert_transpose_kernel(const float* __restrict__ W,
                                         _Float16* __restrict__ Wt,
                                         int K, int N) {
  int idx = blockIdx.x * blockDim.x + threadIdx.x;   // over N * K/4
  int total = N * (K >> 2);
  if (idx >= total) return;
  int n  = idx % N;                                  // coalesced global reads
  int k4 = (idx / N) * 4;
  h4 v;
#pragma unroll
  for (int j = 0; j < 4; ++j) v[j] = (_Float16)W[(size_t)(k4 + j) * N + n];
  *(h4*)&Wt[(size_t)n * K + k4] = v;
}

// ---------------- generic GEMM:  C[MxN] = A[MxK] @ Wt^T[KxN] + bias ---------
// Wt is the f16 TRANSPOSED weight: Wt[n][k], row stride K.
// grid.x = M/128, grid.y = N/64, block = 256 threads (8 waves, 4x2 of 32x32).

__global__ __launch_bounds__(256) void wmma_gemm_bias_kernel(
    const float* __restrict__ A, int lda,
    const _Float16* __restrict__ Wt,
    const float* __restrict__ bias,
    float* __restrict__ C, int ldc, int K) {
  __shared__ _Float16 As[2][128][APAD];
  __shared__ _Float16 Bt[2][64][BPAD];

  const int tid = threadIdx.x;
  const int bm  = blockIdx.x * 128;
  const int bn  = blockIdx.y * 64;
  const int wid = tid >> 5;
  const int wm  = (wid & 3) * 32;
  const int wn  = (wid >> 2) * 32;

  // B async-copy coordinates: 64 rows x 32 halves = 256 x 16B
  const int brow = tid >> 2;           // 0..63 (n within tile)
  const int bcol = (tid & 3) * 8;      // half offset 0/8/16/24

  v8f acc[2][2] = {};

  // ---- prologue: stage tile kt=0 into buffer 0
  {
    async_copy_b128(&Bt[0][brow][bcol], &Wt[(size_t)(bn + brow) * K + bcol]);
    float4 fa[4];
#pragma unroll
    for (int i = 0; i < 4; ++i) {
      const int e = tid + i * 256, r = e >> 3, c = (e & 7) * 4;
      fa[i] = *(const float4*)&A[(size_t)(bm + r) * lda + c];
    }
#pragma unroll
    for (int i = 0; i < 4; ++i) {
      const int e = tid + i * 256, r = e >> 3, c = (e & 7) * 4;
      *(h4*)&As[0][r][c] = cvt4(fa[i]);
    }
    wait_async();
  }
  __syncthreads();

  int buf = 0;
  for (int kt = 0; kt < K; kt += 32) {
    const int nxt = buf ^ 1;
    const bool more = (kt + 32 < K);       // uniform across block
    float4 fa[4];
    if (more) {                            // issue next tile's loads
      async_copy_b128(&Bt[nxt][brow][bcol], &Wt[(size_t)(bn + brow) * K + kt + 32 + bcol]);
#pragma unroll
      for (int i = 0; i < 4; ++i) {
        const int e = tid + i * 256, r = e >> 3, c = (e & 7) * 4;
        fa[i] = *(const float4*)&A[(size_t)(bm + r) * lda + kt + 32 + c];
      }
    }

    v16h af0 = load_a_frag(&As[buf][wm][0]);
    v16h af1 = load_a_frag(&As[buf][wm + 16][0]);
    v16h bf0 = load_bt_frag(&Bt[buf][wn][0]);
    v16h bf1 = load_bt_frag(&Bt[buf][wn + 16][0]);

    acc[0][0] = __builtin_amdgcn_wmma_f32_16x16x32_f16(false, af0, false, bf0, (short)0, acc[0][0], false, false);
    acc[0][1] = __builtin_amdgcn_wmma_f32_16x16x32_f16(false, af0, false, bf1, (short)0, acc[0][1], false, false);
    acc[1][0] = __builtin_amdgcn_wmma_f32_16x16x32_f16(false, af1, false, bf0, (short)0, acc[1][0], false, false);
    acc[1][1] = __builtin_amdgcn_wmma_f32_16x16x32_f16(false, af1, false, bf1, (short)0, acc[1][1], false, false);

    if (more) {                            // convert + store next A tile
#pragma unroll
      for (int i = 0; i < 4; ++i) {
        const int e = tid + i * 256, r = e >> 3, c = (e & 7) * 4;
        *(h4*)&As[nxt][r][c] = cvt4(fa[i]);
      }
      wait_async();                        // B async writes done before barrier
    }
    __syncthreads();
    buf = nxt;
  }

  const int lane = tid & 31;
  const int cl   = lane & 15;
  const int hi   = lane >> 4;
#pragma unroll
  for (int m = 0; m < 2; ++m)
#pragma unroll
    for (int n = 0; n < 2; ++n) {
      const int ccol = bn + wn + n * 16 + cl;
      const float bv = bias ? bias[ccol] : 0.0f;
#pragma unroll
      for (int r = 0; r < 8; ++r) {
        const int crow = bm + wm + m * 16 + hi * 8 + r;
        C[(size_t)crow * ldc + ccol] = acc[m][n][r] + bv;
      }
    }
}

// ---------------- attention scores: S = (Q_h @ K_h^T) / 8 + maskbias --------
// grid = (1024/128, 1024/64, B*H).  Q,K are [2048 x 256]; head h in cols h*64.

__global__ __launch_bounds__(256) void attn_scores_kernel(
    const float* __restrict__ Q, const float* __restrict__ Km,
    const int* __restrict__ mask, int srcset,
    float* __restrict__ S) {
  __shared__ _Float16 As[2][128][APAD];
  __shared__ _Float16 Bt[2][64][BPAD];   // Bt[key][k] -- contiguous along k

  const int tid = threadIdx.x;
  const int bm  = blockIdx.x * 128;    // query base
  const int bn  = blockIdx.y * 64;     // key base
  const int z   = blockIdx.z;          // b*H + h
  const int b   = z >> 2;
  const int h   = z & 3;

  const int wid = tid >> 5;
  const int wm  = (wid & 3) * 32;
  const int wn  = (wid >> 2) * 32;

  const int skey = tid >> 3;           // key column 0..31 (and +32)
  const int skk  = (tid & 7) * 4;      // k offset

  v8f acc[2][2] = {};

  {
    float4 fa[4], fb0, fb1;
#pragma unroll
    for (int i = 0; i < 4; ++i) {
      const int e = tid + i * 256, r = e >> 3, c = (e & 7) * 4;
      fa[i] = *(const float4*)&Q[(size_t)(b * NSEQ + bm + r) * DMODEL + h * DHEAD + c];
    }
    fb0 = *(const float4*)&Km[(size_t)(b * NSEQ + bn + skey)      * DMODEL + h * DHEAD + skk];
    fb1 = *(const float4*)&Km[(size_t)(b * NSEQ + bn + skey + 32) * DMODEL + h * DHEAD + skk];
#pragma unroll
    for (int i = 0; i < 4; ++i) {
      const int e = tid + i * 256, r = e >> 3, c = (e & 7) * 4;
      *(h4*)&As[0][r][c] = cvt4(fa[i]);
    }
    *(h4*)&Bt[0][skey][skk]      = cvt4(fb0);
    *(h4*)&Bt[0][skey + 32][skk] = cvt4(fb1);
  }
  __syncthreads();

#pragma unroll
  for (int kt = 0; kt < DHEAD; kt += 32) {
    const int buf = (kt >> 5) & 1;
    const int nxt = buf ^ 1;
    const bool more = (kt + 32 < DHEAD);
    float4 fa[4], fb0, fb1;
    if (more) {
#pragma unroll
      for (int i = 0; i < 4; ++i) {
        const int e = tid + i * 256, r = e >> 3, c = (e & 7) * 4;
        fa[i] = *(const float4*)&Q[(size_t)(b * NSEQ + bm + r) * DMODEL + h * DHEAD + kt + 32 + c];
      }
      fb0 = *(const float4*)&Km[(size_t)(b * NSEQ + bn + skey)      * DMODEL + h * DHEAD + kt + 32 + skk];
      fb1 = *(const float4*)&Km[(size_t)(b * NSEQ + bn + skey + 32) * DMODEL + h * DHEAD + kt + 32 + skk];
    }

    v16h af0 = load_a_frag(&As[buf][wm][0]);
    v16h af1 = load_a_frag(&As[buf][wm + 16][0]);
    v16h bf0 = load_bt_frag(&Bt[buf][wn][0]);
    v16h bf1 = load_bt_frag(&Bt[buf][wn + 16][0]);

    acc[0][0] = __builtin_amdgcn_wmma_f32_16x16x32_f16(false, af0, false, bf0, (short)0, acc[0][0], false, false);
    acc[0][1] = __builtin_amdgcn_wmma_f32_16x16x32_f16(false, af0, false, bf1, (short)0, acc[0][1], false, false);
    acc[1][0] = __builtin_amdgcn_wmma_f32_16x16x32_f16(false, af1, false, bf0, (short)0, acc[1][0], false, false);
    acc[1][1] = __builtin_amdgcn_wmma_f32_16x16x32_f16(false, af1, false, bf1, (short)0, acc[1][1], false, false);

    if (more) {
#pragma unroll
      for (int i = 0; i < 4; ++i) {
        const int e = tid + i * 256, r = e >> 3, c = (e & 7) * 4;
        *(h4*)&As[nxt][r][c] = cvt4(fa[i]);
      }
      *(h4*)&Bt[nxt][skey][skk]      = cvt4(fb0);
      *(h4*)&Bt[nxt][skey + 32][skk] = cvt4(fb1);
    }
    __syncthreads();
  }

  const int lane = tid & 31;
  const int cl   = lane & 15;
  const int hi   = lane >> 4;
#pragma unroll
  for (int m = 0; m < 2; ++m)
#pragma unroll
    for (int n = 0; n < 2; ++n) {
      const int key = bn + wn + n * 16 + cl;
      const float mb = (mask[(b * 2 + srcset) * NSEQ + key] == 0) ? -1.0e9f : 0.0f;
#pragma unroll
      for (int r = 0; r < 8; ++r) {
        const int qi = bm + wm + m * 16 + hi * 8 + r;
        S[((size_t)z * NSEQ + qi) * NSEQ + key] = acc[m][n][r] * 0.125f + mb;
      }
    }
}

// ---------------- softmax over 1024-length rows ----------------------------

__global__ __launch_bounds__(256) void softmax_kernel(float* __restrict__ S) {
  __shared__ float red[256];
  float* p = S + (size_t)blockIdx.x * NSEQ;
  const int tid = threadIdx.x;

  float m = -3.4e38f;
#pragma unroll
  for (int i = 0; i < 4; ++i) m = fmaxf(m, p[tid + i * 256]);
  red[tid] = m; __syncthreads();
  for (int s = 128; s > 0; s >>= 1) { if (tid < s) red[tid] = fmaxf(red[tid], red[tid + s]); __syncthreads(); }
  m = red[0]; __syncthreads();

  float sum = 0.0f;
#pragma unroll
  for (int i = 0; i < 4; ++i) { float e = __expf(p[tid + i * 256] - m); p[tid + i * 256] = e; sum += e; }
  red[tid] = sum; __syncthreads();
  for (int s = 128; s > 0; s >>= 1) { if (tid < s) red[tid] += red[tid + s]; __syncthreads(); }
  const float inv = 1.0f / red[0];
#pragma unroll
  for (int i = 0; i < 4; ++i) p[tid + i * 256] *= inv;
}

// ---------------- attention output: O_h = P @ V_h ---------------------------
// grid = (1024/128, 1, B*H).  K = 1024, N = 64.

__global__ __launch_bounds__(256) void attn_pv_kernel(
    const float* __restrict__ P, const float* __restrict__ V,
    float* __restrict__ O) {
  __shared__ _Float16 As[2][128][APAD];
  __shared__ _Float16 Bt[2][64][BPAD];

  const int tid = threadIdx.x;
  const int bm  = blockIdx.x * 128;
  const int z   = blockIdx.z;
  const int b   = z >> 2;
  const int h   = z & 3;

  const int wid = tid >> 5;
  const int wm  = (wid & 3) * 32;
  const int wn  = (wid >> 2) * 32;

  const int br = (tid >> 4) * 2;       // key k-pair
  const int bc = (tid & 15) * 4;       // head col quad

  v8f acc[2][2] = {};

  {
    float4 fa[4], fb0, fb1;
#pragma unroll
    for (int i = 0; i < 4; ++i) {
      const int e = tid + i * 256, r = e >> 3, c = (e & 7) * 4;
      fa[i] = *(const float4*)&P[((size_t)z * NSEQ + bm + r) * NSEQ + c];
    }
    fb0 = *(const float4*)&V[(size_t)(b * NSEQ + br)     * DMODEL + h * DHEAD + bc];
    fb1 = *(const float4*)&V[(size_t)(b * NSEQ + br + 1) * DMODEL + h * DHEAD + bc];
#pragma unroll
    for (int i = 0; i < 4; ++i) {
      const int e = tid + i * 256, r = e >> 3, c = (e & 7) * 4;
      *(h4*)&As[0][r][c] = cvt4(fa[i]);
    }
    store_bpair(&Bt[0][bc][br], fb0, fb1);
  }
  __syncthreads();

  int buf = 0;
  for (int kt = 0; kt < NSEQ; kt += 32) {
    const int nxt = buf ^ 1;
    const bool more = (kt + 32 < NSEQ);
    float4 fa[4], fb0, fb1;
    if (more) {
#pragma unroll
      for (int i = 0; i < 4; ++i) {
        const int e = tid + i * 256, r = e >> 3, c = (e & 7) * 4;
        fa[i] = *(const float4*)&P[((size_t)z * NSEQ + bm + r) * NSEQ + kt + 32 + c];
      }
      fb0 = *(const float4*)&V[(size_t)(b * NSEQ + kt + 32 + br)     * DMODEL + h * DHEAD + bc];
      fb1 = *(const float4*)&V[(size_t)(b * NSEQ + kt + 32 + br + 1) * DMODEL + h * DHEAD + bc];
    }

    v16h af0 = load_a_frag(&As[buf][wm][0]);
    v16h af1 = load_a_frag(&As[buf][wm + 16][0]);
    v16h bf0 = load_bt_frag(&Bt[buf][wn][0]);
    v16h bf1 = load_bt_frag(&Bt[buf][wn + 16][0]);

    acc[0][0] = __builtin_amdgcn_wmma_f32_16x16x32_f16(false, af0, false, bf0, (short)0, acc[0][0], false, false);
    acc[0][1] = __builtin_amdgcn_wmma_f32_16x16x32_f16(false, af0, false, bf1, (short)0, acc[0][1], false, false);
    acc[1][0] = __builtin_amdgcn_wmma_f32_16x16x32_f16(false, af1, false, bf0, (short)0, acc[1][0], false, false);
    acc[1][1] = __builtin_amdgcn_wmma_f32_16x16x32_f16(false, af1, false, bf1, (short)0, acc[1][1], false, false);

    if (more) {
#pragma unroll
      for (int i = 0; i < 4; ++i) {
        const int e = tid + i * 256, r = e >> 3, c = (e & 7) * 4;
        *(h4*)&As[nxt][r][c] = cvt4(fa[i]);
      }
      store_bpair(&Bt[nxt][bc][br], fb0, fb1);
    }
    __syncthreads();
    buf = nxt;
  }

  const int lane = tid & 31;
  const int cl   = lane & 15;
  const int hi   = lane >> 4;
#pragma unroll
  for (int m = 0; m < 2; ++m)
#pragma unroll
    for (int n = 0; n < 2; ++n) {
      const int oc = wn + n * 16 + cl;       // 0..63 within head
#pragma unroll
      for (int r = 0; r < 8; ++r) {
        const int qi = bm + wm + m * 16 + hi * 8 + r;
        O[(size_t)(b * NSEQ + qi) * DMODEL + h * DHEAD + oc] = acc[m][n][r];
      }
    }
}

// ---------------- elementwise / reduction kernels ---------------------------

__global__ void slice_copy_kernel(const float* __restrict__ src, float* __restrict__ dst, int s) {
  int idx = blockIdx.x * blockDim.x + threadIdx.x;      // over 2048*256
  if (idx >= NROWS * DMODEL) return;
  int b = idx / (NSEQ * DMODEL);
  int rem = idx - b * (NSEQ * DMODEL);
  dst[idx] = src[(size_t)(b * 2 + s) * NSEQ * DMODEL + rem];
}

__global__ void pack_out_kernel(const float* __restrict__ x, float* __restrict__ out, int s) {
  int idx = blockIdx.x * blockDim.x + threadIdx.x;
  if (idx >= NROWS * DMODEL) return;
  int b = idx / (NSEQ * DMODEL);
  int rem = idx - b * (NSEQ * DMODEL);
  out[(size_t)(b * 2 + s) * NSEQ * DMODEL + rem] = x[idx];
}

__global__ void concat_kernel(const float* __restrict__ x, const float* __restrict__ a,
                              float* __restrict__ hcat) {
  int idx = blockIdx.x * blockDim.x + threadIdx.x;      // over 2048*512
  if (idx >= NROWS * DHID) return;
  int row = idx >> 9;
  int col = idx & 511;
  hcat[idx] = (col < DMODEL) ? x[row * DMODEL + col] : a[row * DMODEL + col - DMODEL];
}

__global__ __launch_bounds__(256) void bn_reduce_kernel(const float* __restrict__ H,
                                                        float* __restrict__ mv) {
  __shared__ float rs[256], rss[256];
  const int col = blockIdx.x;                            // 0..511
  const int tid = threadIdx.x;
  float s = 0.0f, ss = 0.0f;
#pragma unroll
  for (int i = 0; i < 8; ++i) {
    float v = H[(size_t)(tid + i * 256) * DHID + col];
    s += v; ss += v * v;
  }
  rs[tid] = s; rss[tid] = ss; __syncthreads();
  for (int k = 128; k > 0; k >>= 1) {
    if (tid < k) { rs[tid] += rs[tid + k]; rss[tid] += rss[tid + k]; }
    __syncthreads();
  }
  if (tid == 0) {
    float mean = rs[0] * (1.0f / (float)NROWS);
    float var  = rss[0] * (1.0f / (float)NROWS) - mean * mean;
    mv[col] = mean;
    mv[DHID + col] = rsqrtf(var + 1e-5f);
  }
}

__global__ void bn_apply_relu_kernel(float* __restrict__ H, const float* __restrict__ mv,
                                     const float* __restrict__ g, const float* __restrict__ be) {
  int idx = blockIdx.x * blockDim.x + threadIdx.x;      // over 2048*512
  if (idx >= NROWS * DHID) return;
  int col = idx & 511;
  float v = (H[idx] - mv[col]) * mv[DHID + col] * g[col] + be[col];
  H[idx] = v > 0.0f ? v : 0.0f;
}

__global__ void add_kernel(float* __restrict__ x, const float* __restrict__ d, int n) {
  int idx = blockIdx.x * blockDim.x + threadIdx.x;
  if (idx < n) x[idx] += d[idx];
}

// ---------------------------------------------------------------------------

extern "C" void kernel_launch(void* const* d_in, const int* in_sizes, int n_in,
                              void* d_out, int out_size, void* d_ws, size_t ws_size,
                              hipStream_t stream) {
  (void)in_sizes; (void)n_in; (void)out_size; (void)ws_size;

  const float* desc = (const float*)d_in[0];
  const int*   mask = (const int*)d_in[1];
  const float* qw = (const float*)d_in[2];  const float* qb = (const float*)d_in[3];
  const float* kw = (const float*)d_in[4];  const float* kb = (const float*)d_in[5];
  const float* vw = (const float*)d_in[6];  const float* vb = (const float*)d_in[7];
  const float* ow = (const float*)d_in[8];  const float* ob = (const float*)d_in[9];
  const float* w1 = (const float*)d_in[10]; const float* b1 = (const float*)d_in[11];
  const float* bng = (const float*)d_in[12]; const float* bnb = (const float*)d_in[13];
  const float* w2 = (const float*)d_in[14]; const float* b2 = (const float*)d_in[15];

  const size_t XSZ = (size_t)NROWS * DMODEL;      // 524288
  const size_t HSZ = (size_t)NROWS * DHID;        // 1048576
  const size_t SSZ = (size_t)2 * NHEAD * NSEQ * NSEQ;   // 8M floats
  float* ws   = (float*)d_ws;
  float* x0   = ws;           float* x1   = x0 + XSZ;
  float* qB   = x1 + XSZ;     float* kB   = qB + XSZ;
  float* vB   = kB + XSZ;     float* attn = vB + XSZ;
  float* aP   = attn + XSZ;
  float* hcat = aP + XSZ;     float* hlin = hcat + HSZ;
  float* dl0  = hlin + HSZ;   float* dl1  = dl0 + XSZ;
  float* mv   = dl1 + XSZ;    float* S    = mv + 2 * DHID;

  // f16 transposed weights, per-layer block:
  //   [qw kw vw ow](4 x 256*256) [w1](512*512) [w2](256*512)
  _Float16* wt = (_Float16*)(S + SSZ);
  const size_t WSQ  = (size_t)DMODEL * DMODEL;    // 65536
  const size_t W1SZ = (size_t)DHID * DHID;        // 262144
  const size_t W2SZ = (size_t)DMODEL * DHID;      // 131072
  const size_t WTL  = 4 * WSQ + W1SZ + W2SZ;      // per-layer halves

  const dim3 blk(256);
  const int EB_X = (int)((XSZ + 255) / 256);
  const int EB_H = (int)((HSZ + 255) / 256);

  // ---- one-time (per call) weight convert + transpose ----
  for (int i = 0; i < 18; ++i) {
    _Float16* wl = wt + (size_t)i * WTL;
    convert_transpose_kernel<<<64, blk, 0, stream>>>(qw + i * WSQ, wl,            DMODEL, DMODEL);
    convert_transpose_kernel<<<64, blk, 0, stream>>>(kw + i * WSQ, wl + WSQ,      DMODEL, DMODEL);
    convert_transpose_kernel<<<64, blk, 0, stream>>>(vw + i * WSQ, wl + 2 * WSQ,  DMODEL, DMODEL);
    convert_transpose_kernel<<<64, blk, 0, stream>>>(ow + i * WSQ, wl + 3 * WSQ,  DMODEL, DMODEL);
    convert_transpose_kernel<<<256, blk, 0, stream>>>(w1 + i * W1SZ, wl + 4 * WSQ, DHID, DHID);
    convert_transpose_kernel<<<128, blk, 0, stream>>>(w2 + i * W2SZ, wl + 4 * WSQ + W1SZ, DHID, DMODEL);
  }

  slice_copy_kernel<<<EB_X, blk, 0, stream>>>(desc, x0, 0);
  slice_copy_kernel<<<EB_X, blk, 0, stream>>>(desc, x1, 1);

  for (int i = 0; i < 18; ++i) {
    const int cross = i & 1;                           // 'self','cross' alternating
    _Float16* wl = wt + (size_t)i * WTL;
    for (int s = 0; s < 2; ++s) {
      float* x   = s ? x1 : x0;
      float* src = cross ? (s ? x0 : x1) : x;
      const int srcset = cross ? (1 - s) : s;
      float* dl  = s ? dl1 : dl0;

      // Q/K/V projections: [2048x256] @ [256x256] + bias
      wmma_gemm_bias_kernel<<<dim3(16, 4), blk, 0, stream>>>(
          x,   DMODEL, wl,           qb + i * DMODEL, qB, DMODEL, DMODEL);
      wmma_gemm_bias_kernel<<<dim3(16, 4), blk, 0, stream>>>(
          src, DMODEL, wl + WSQ,     kb + i * DMODEL, kB, DMODEL, DMODEL);
      wmma_gemm_bias_kernel<<<dim3(16, 4), blk, 0, stream>>>(
          src, DMODEL, wl + 2 * WSQ, vb + i * DMODEL, vB, DMODEL, DMODEL);

      // attention
      attn_scores_kernel<<<dim3(8, 16, 8), blk, 0, stream>>>(qB, kB, mask, srcset, S);
      softmax_kernel<<<2 * NHEAD * NSEQ, blk, 0, stream>>>(S);
      attn_pv_kernel<<<dim3(8, 1, 8), blk, 0, stream>>>(S, vB, attn);

      // output projection
      wmma_gemm_bias_kernel<<<dim3(16, 4), blk, 0, stream>>>(
          attn, DMODEL, wl + 3 * WSQ, ob + i * DMODEL, aP, DMODEL, DMODEL);

      // MLP: concat -> 512x512 -> BN -> ReLU -> 512x256
      concat_kernel<<<EB_H, blk, 0, stream>>>(x, aP, hcat);
      wmma_gemm_bias_kernel<<<dim3(16, 8), blk, 0, stream>>>(
          hcat, DHID, wl + 4 * WSQ, b1 + i * DHID, hlin, DHID, DHID);
      bn_reduce_kernel<<<DHID, blk, 0, stream>>>(hlin, mv);
      bn_apply_relu_kernel<<<EB_H, blk, 0, stream>>>(hlin, mv, bng + i * DHID, bnb + i * DHID);
      wmma_gemm_bias_kernel<<<dim3(16, 4), blk, 0, stream>>>(
          hlin, DHID, wl + 4 * WSQ + W1SZ, b2 + i * DMODEL, dl, DMODEL, DHID);
    }
    // residual updates after BOTH deltas are computed (cross-attn reads pre-update state)
    add_kernel<<<EB_X, blk, 0, stream>>>(x0, dl0, (int)XSZ);
    add_kernel<<<EB_X, blk, 0, stream>>>(x1, dl1, (int)XSZ);
  }

  pack_out_kernel<<<EB_X, blk, 0, stream>>>(x0, (float*)d_out, 0);
  pack_out_kernel<<<EB_X, blk, 0, stream>>>(x1, (float*)d_out, 1);
}
